// SwinBlock_63058709840161
// MI455X (gfx1250) — matrix operand order repaired and checked by reading, not compiled, not run on hardware
//
#include <hip/hip_runtime.h>
#include <cstdint>

// ---------------- constants (match reference) ----------------
#define K_B     8
#define K_H     64
#define K_W     64
#define K_C     512
#define K_WS    8
#define K_SHIFT 8
#define K_HEADS 16
#define K_HD    32
#define K_E4    2048
#define K_TOK   (K_B * K_H * K_W)     /* 32768 */
#define K_WT    (K_WS * K_WS)         /* 64 tokens / window */
#define K_NWIN  (K_TOK / K_WT)        /* 512 windows */

typedef __bf16 bf16;
typedef __attribute__((ext_vector_type(16))) __bf16 v16bf;
typedef __attribute__((ext_vector_type(8)))  __bf16 v8bf;
typedef __attribute__((ext_vector_type(8)))  float  v8f;

union BF16x16 { v16bf v; v8bf h[2]; };

// ---------------- WMMA helpers ----------------
__device__ __forceinline__ v8f wmma_bf16(v16bf a, v16bf b, v8f c) {
  // D = A(16x32) * B(32x16) + C(16x16), f32 accumulate
  return __builtin_amdgcn_wmma_f32_16x16x32_bf16(false, a, false, b, (short)0, c, false, false);
}

// A fragment (16x32, row-major source, leading dim ld):
// lanes 0-15 row=lane:  a[0..15]=K0..15
// lanes16-31 row=l-16:  a[0..15]=K16..31
__device__ __forceinline__ v16bf load_a_frag(const bf16* base, int ld) {
  int lane = threadIdx.x & 31;
  int half = lane >> 4, row = lane & 15;
  const bf16* p = base + (size_t)row * ld + half * 16;
  BF16x16 f;
  f.h[0] = *(const v8bf*)(p);
  f.h[1] = *(const v8bf*)(p + 8);
  return f.v;
}

// B fragment from pre-packed weights: 32 lanes x 16 bf16 contiguous per tile.
__device__ __forceinline__ v16bf load_b_packed(const bf16* tile) {
  int lane = threadIdx.x & 31;
  const bf16* p = tile + (size_t)lane * 16;
  BF16x16 f;
  f.h[0] = *(const v8bf*)(p);
  f.h[1] = *(const v8bf*)(p + 8);
  return f.v;
}

// B fragment where column n's K-vector is contiguous (ld = column stride):
// lanes 0-15: col=lane,   b[e] = col[k0 + e]
// lanes16-31: col=lane-16 b[e] = col[k0 + 16 + e]
__device__ __forceinline__ v16bf load_b_cols(const bf16* base, int ld, int k0) {
  int lane = threadIdx.x & 31;
  int half = lane >> 4, col = lane & 15;
  const bf16* p = base + (size_t)col * ld + k0 + half * 16;
  BF16x16 f;
  f.h[0] = *(const v8bf*)(p);
  f.h[1] = *(const v8bf*)(p + 8);
  return f.v;
}

// ---------------- CDNA5 async global->LDS staging ----------------
// GLOBAL_LOAD_ASYNC_TO_LDS_B128: memory->LDS without VGPR data, tracked by
// ASYNCcnt (CDNA5 ISA). Flat LDS addresses carry the
// wave-relative LDS byte offset in the low 32 bits (aperture mapping, §10.2),
// so truncating the generic shared pointer yields the VDST operand.
__device__ __forceinline__ void async_ld_b128(void* lds_dst, const void* gsrc) {
  uint32_t l = (uint32_t)(uintptr_t)lds_dst;
  asm volatile("global_load_async_to_lds_b128 %0, %1, off"
               :
               : "v"(l), "v"(gsrc)
               : "memory");
}
__device__ __forceinline__ void wait_async0() {
  asm volatile("s_wait_asynccnt 0x0" ::: "memory");
}

// Stage `bytes` (multiple of 16*256) from gsrc to LDS with all 256 threads.
__device__ __forceinline__ void stage_tile(void* lds, const void* gsrc, int bytes) {
  char* d = (char*)lds;
  const char* s = (const char*)gsrc;
  for (int i = threadIdx.x; i < bytes / 16; i += 256)
    async_ld_b128(d + i * 16, s + i * 16);
  wait_async0();
  __syncthreads();
}

// ---------------- K1: pack weights (f32 KxN row-major -> bf16 B-fragment order) ----------------
__global__ void pack_w(const float* __restrict__ w, bf16* __restrict__ out, int K, int N) {
  int ntiles = N >> 4;
  int ktiles = K >> 5;
  int total = ktiles * ntiles * 32;           // one (tile,lane) per index
  for (int i = blockIdx.x * blockDim.x + threadIdx.x; i < total;
       i += gridDim.x * blockDim.x) {
    int lane = i & 31;
    int tile = i >> 5;
    int nt = tile % ntiles;
    int kt = tile / ntiles;
    int half = lane >> 4;
    int col = nt * 16 + (lane & 15);
    int kbase = kt * 32 + half * 16;
    bf16* o = out + (size_t)i * 16;
#pragma unroll
    for (int e = 0; e < 16; ++e)
      o[e] = (bf16)w[(size_t)(kbase + e) * N + col];
  }
}

// ---------------- K2: roll(-8,-8) + LN1 + window partition -> bf16 ----------------
__global__ void __launch_bounds__(256) ln1_win(const float* __restrict__ x,
                                               const float* __restrict__ g,
                                               const float* __restrict__ b,
                                               bf16* __restrict__ winx) {
  int wid  = (blockIdx.x * blockDim.x + threadIdx.x) >> 5;  // one wave per token
  int lane = threadIdx.x & 31;
  if (wid >= K_TOK) return;
  int win = wid >> 6, tok = wid & 63;
  int bi = win >> 6;
  int wy = (win >> 3) & 7, wx = win & 7;
  int iy = tok >> 3, ix = tok & 7;
  int sy = (wy * 8 + iy + K_SHIFT) & 63;
  int sx = (wx * 8 + ix + K_SHIFT) & 63;
  const float* row = x + ((size_t)bi * 4096 + sy * 64 + sx) * K_C;

  float vals[16];
  float s = 0.f, s2 = 0.f;
#pragma unroll
  for (int i = 0; i < 16; ++i) {
    float v = row[lane + i * 32];
    vals[i] = v; s += v; s2 += v * v;
  }
#pragma unroll
  for (int off = 16; off >= 1; off >>= 1) {
    s  += __shfl_xor(s,  off, 32);
    s2 += __shfl_xor(s2, off, 32);
  }
  float mean = s * (1.f / K_C);
  float var  = s2 * (1.f / K_C) - mean * mean;
  float inv  = rsqrtf(var + 1e-5f);
  bf16* o = winx + (size_t)wid * K_C;
#pragma unroll
  for (int i = 0; i < 16; ++i) {
    int c = lane + i * 32;
    o[c] = (bf16)((vals[i] - mean) * inv * g[c] + b[c]);
  }
}

// ---------------- K3: QKV GEMM (per-window 64x1536x512) ----------------
__global__ void __launch_bounds__(256) qkv_gemm(const bf16* __restrict__ winx,
                                                const bf16* __restrict__ wp,
                                                const float* __restrict__ bias,
                                                bf16* __restrict__ qb,
                                                bf16* __restrict__ kb,
                                                bf16* __restrict__ vtb) {
  __shared__ bf16 atile[K_WT * K_C];   // 64x512 bf16 = 64 KB
  int win = blockIdx.x;
  stage_tile(atile, winx + (size_t)win * K_WT * K_C, K_WT * K_C * 2);

  int wave = threadIdx.x >> 5;
  int lane = threadIdx.x & 31;
  int half = lane >> 4, l15 = lane & 15;

  for (int mt = 0; mt < 4; ++mt) {
    for (int nt0 = wave; nt0 < 96; nt0 += 8) {
      // nt / head / which are wave-uniform: force them scalar so the q|k|v
      // scatter compiles to s_cbranch instead of exec-mask divergence.
      int nt    = __builtin_amdgcn_readfirstlane(nt0);
      v8f acc = {};
#pragma unroll 4
      for (int kt = 0; kt < 16; ++kt) {
        // keep the B fragment two K-steps ahead warm (global_prefetch_b8)
        __builtin_prefetch(wp + (size_t)((kt + 2) * 96 + nt) * 32 * 16, 0, 0);
        v16bf a = load_a_frag(atile + mt * 16 * K_C + kt * 32, K_C);
        v16bf bb = load_b_packed(wp + (size_t)(kt * 96 + nt) * 32 * 16);
        acc = wmma_bf16(a, bb, acc);
      }
      int head  = nt / 6;                 // scalar division
      int g6    = nt - head * 6;          // 0..5
      int which = g6 >> 1;                // 0=q,1=k,2=v (scalar)
      int hd    = (g6 & 1) * 16 + l15;    // 0..31
      float bv  = bias[nt * 16 + l15];
      size_t hb = (size_t)win * K_HEADS + head;
      if (which == 0) {
#pragma unroll
        for (int r = 0; r < 8; ++r) {
          int tok = mt * 16 + half * 8 + r;
          qb[(hb * K_WT + tok) * K_HD + hd] = (bf16)(acc[r] + bv);
        }
      } else if (which == 1) {
#pragma unroll
        for (int r = 0; r < 8; ++r) {
          int tok = mt * 16 + half * 8 + r;
          kb[(hb * K_WT + tok) * K_HD + hd] = (bf16)(acc[r] + bv);
        }
      } else {
#pragma unroll
        for (int r = 0; r < 8; ++r) {
          int tok = mt * 16 + half * 8 + r;
          vtb[(hb * K_HD + hd) * K_WT + tok] = (bf16)(acc[r] + bv);  // V transposed
        }
      }
    }
  }
}

// ---------------- K4: attention (one wave per window*head) ----------------
__global__ void __launch_bounds__(256) attn_kernel(const bf16* __restrict__ qb,
                                                   const bf16* __restrict__ kb,
                                                   const bf16* __restrict__ vtb,
                                                   bf16* __restrict__ attn_o) {
  __shared__ bf16 pbuf_all[8 * 16 * 64];   // per-wave 16x64 bf16 P staging
  int wave = threadIdx.x >> 5;
  int lane = threadIdx.x & 31;
  int half = lane >> 4, l15 = lane & 15;
  int gid  = blockIdx.x * 8 + wave;        // 0..8191
  int win  = gid >> 4, head = gid & 15;

  size_t hb = (size_t)win * K_HEADS + head;
  const bf16* Q = qb  + hb * K_WT * K_HD;
  const bf16* K = kb  + hb * K_WT * K_HD;
  const bf16* V = vtb + hb * K_HD * K_WT;
  bf16* O = attn_o + (size_t)win * K_WT * K_C + head * K_HD;
  bf16* pbuf = pbuf_all + wave * 16 * 64;

  // K fragments for S = Q*K^T : B[k=hd][n=ktok], columns (tokens) contiguous
  v16bf kfrag[4];
#pragma unroll
  for (int nt = 0; nt < 4; ++nt)
    kfrag[nt] = load_b_cols(K + (size_t)(nt * 16) * K_HD, K_HD, 0);

  // V fragments for O = P*V : B[k=tok][n=hd], vT makes columns contiguous
  v16bf vfrag[2][2];
#pragma unroll
  for (int kt = 0; kt < 2; ++kt)
#pragma unroll
    for (int nc = 0; nc < 2; ++nc)
      vfrag[kt][nc] = load_b_cols(V + (size_t)(nc * 16) * K_WT, K_WT, kt * 32);

  const float scale = 0.17677669529663687f;  // 1/sqrt(32)

  for (int mt = 0; mt < 4; ++mt) {
    v16bf qfrag = load_a_frag(Q + (size_t)(mt * 16) * K_HD, K_HD);
    v8f s[4];
#pragma unroll
    for (int nt = 0; nt < 4; ++nt) {
      v8f z = {};
      s[nt] = wmma_bf16(qfrag, kfrag[nt], z);
    }
    // exact row softmax; row = mt*16 + half*8 + r, cols spread over (nt, lane&15)
#pragma unroll
    for (int r = 0; r < 8; ++r) {
      float m = fmaxf(fmaxf(s[0][r], s[1][r]), fmaxf(s[2][r], s[3][r]));
#pragma unroll
      for (int off = 1; off < 16; off <<= 1) m = fmaxf(m, __shfl_xor(m, off, 32));
      float e0 = __expf((s[0][r] - m) * scale);
      float e1 = __expf((s[1][r] - m) * scale);
      float e2 = __expf((s[2][r] - m) * scale);
      float e3 = __expf((s[3][r] - m) * scale);
      float sum = e0 + e1 + e2 + e3;
#pragma unroll
      for (int off = 1; off < 16; off <<= 1) sum += __shfl_xor(sum, off, 32);
      float rinv = 1.f / sum;
      int row = half * 8 + r;
      pbuf[row * 64 +  0 + l15] = (bf16)(e0 * rinv);
      pbuf[row * 64 + 16 + l15] = (bf16)(e1 * rinv);
      pbuf[row * 64 + 32 + l15] = (bf16)(e2 * rinv);
      pbuf[row * 64 + 48 + l15] = (bf16)(e3 * rinv);
    }
    // O(16x32) = P(16x64) @ V(64x32)
    v8f o0 = {}, o1 = {};
#pragma unroll
    for (int kt = 0; kt < 2; ++kt) {
      v16bf pf = load_a_frag(pbuf + kt * 32, 64);
      o0 = wmma_bf16(pf, vfrag[kt][0], o0);
      o1 = wmma_bf16(pf, vfrag[kt][1], o1);
    }
#pragma unroll
    for (int r = 0; r < 8; ++r) {
      int tok = mt * 16 + half * 8 + r;
      O[(size_t)tok * K_C +  0 + l15] = (bf16)o0[r];
      O[(size_t)tok * K_C + 16 + l15] = (bf16)o1[r];
    }
  }
}

// ---------------- K5: proj GEMM + reverse shift/roll scatter + residual ----------------
__global__ void __launch_bounds__(256) proj_gemm(const bf16* __restrict__ attn_o,
                                                 const bf16* __restrict__ wp,
                                                 const float* __restrict__ bias,
                                                 const float* __restrict__ x,
                                                 float* __restrict__ msa) {
  __shared__ bf16 atile[K_WT * K_C];   // 64 KB
  int win = blockIdx.x;
  stage_tile(atile, attn_o + (size_t)win * K_WT * K_C, K_WT * K_C * 2);

  int wave = threadIdx.x >> 5;
  int lane = threadIdx.x & 31;
  int half = lane >> 4, l15 = lane & 15;
  int bi = win >> 6, wy = (win >> 3) & 7, wx = win & 7;

  for (int mt = 0; mt < 4; ++mt) {
    for (int nt0 = wave; nt0 < 32; nt0 += 8) {
      int nt = __builtin_amdgcn_readfirstlane(nt0);
      v8f acc = {};
#pragma unroll 4
      for (int kt = 0; kt < 16; ++kt) {
        __builtin_prefetch(wp + (size_t)((kt + 2) * 32 + nt) * 32 * 16, 0, 0);
        v16bf a = load_a_frag(atile + mt * 16 * K_C + kt * 32, K_C);
        v16bf bb = load_b_packed(wp + (size_t)(kt * 32 + nt) * 32 * 16);
        acc = wmma_bf16(a, bb, acc);
      }
      int ncol = nt * 16 + l15;
      float bv = bias[ncol];
#pragma unroll
      for (int r = 0; r < 8; ++r) {
        int tok = mt * 16 + half * 8 + r;
        int iy = tok >> 3, ix = tok & 7;
        int y = (wy * 8 + iy + K_SHIFT) & 63;
        int xx = (wx * 8 + ix + K_SHIFT) & 63;
        size_t row = (size_t)bi * 4096 + y * 64 + xx;
        msa[row * K_C + ncol] = acc[r] + bv + x[row * K_C + ncol];
      }
    }
  }
}

// ---------------- K6: LN2 (wave per row) ----------------
__global__ void __launch_bounds__(256) ln2_kernel(const float* __restrict__ msa,
                                                  const float* __restrict__ g,
                                                  const float* __restrict__ b,
                                                  bf16* __restrict__ out) {
  int wid  = (blockIdx.x * blockDim.x + threadIdx.x) >> 5;
  int lane = threadIdx.x & 31;
  if (wid >= K_TOK) return;
  const float* row = msa + (size_t)wid * K_C;
  float vals[16];
  float s = 0.f, s2 = 0.f;
#pragma unroll
  for (int i = 0; i < 16; ++i) {
    float v = row[lane + i * 32];
    vals[i] = v; s += v; s2 += v * v;
  }
#pragma unroll
  for (int off = 16; off >= 1; off >>= 1) {
    s  += __shfl_xor(s,  off, 32);
    s2 += __shfl_xor(s2, off, 32);
  }
  float mean = s * (1.f / K_C);
  float var  = s2 * (1.f / K_C) - mean * mean;
  float inv  = rsqrtf(var + 1e-5f);
  bf16* o = out + (size_t)wid * K_C;
#pragma unroll
  for (int i = 0; i < 16; ++i) {
    int c = lane + i * 32;
    o[c] = (bf16)((vals[i] - mean) * inv * g[c] + b[c]);
  }
}

// ---------------- K7: MLP up-proj + exact GELU ----------------
__global__ void __launch_bounds__(256) mlp1_gemm(const bf16* __restrict__ ln2x,
                                                 const bf16* __restrict__ wp,
                                                 const float* __restrict__ bias,
                                                 bf16* __restrict__ h) {
  __shared__ bf16 atile[K_WT * K_C];   // 64 rows x 512
  int blk = blockIdx.x;                // 512 blocks of 64 rows
  stage_tile(atile, ln2x + (size_t)blk * K_WT * K_C, K_WT * K_C * 2);

  int wave = threadIdx.x >> 5;
  int lane = threadIdx.x & 31;
  int half = lane >> 4, l15 = lane & 15;

  for (int mt = 0; mt < 4; ++mt) {
    for (int nt0 = wave; nt0 < 128; nt0 += 8) {
      int nt = __builtin_amdgcn_readfirstlane(nt0);
      v8f acc = {};
#pragma unroll 4
      for (int kt = 0; kt < 16; ++kt) {
        __builtin_prefetch(wp + (size_t)((kt + 2) * 128 + nt) * 32 * 16, 0, 0);
        v16bf a = load_a_frag(atile + mt * 16 * K_C + kt * 32, K_C);
        v16bf bb = load_b_packed(wp + (size_t)(kt * 128 + nt) * 32 * 16);
        acc = wmma_bf16(a, bb, acc);
      }
      int ncol = nt * 16 + l15;
      float bv = bias[ncol];
#pragma unroll
      for (int r = 0; r < 8; ++r) {
        int row = blk * 64 + mt * 16 + half * 8 + r;
        float u = acc[r] + bv;
        float ge = 0.5f * u * (1.f + erff(u * 0.70710678118654752f));
        h[(size_t)row * K_E4 + ncol] = (bf16)ge;
      }
    }
  }
}

// ---------------- K8: MLP down-proj + residual -> output ----------------
__global__ void __launch_bounds__(256) mlp2_gemm(const bf16* __restrict__ h,
                                                 const bf16* __restrict__ wp,
                                                 const float* __restrict__ bias,
                                                 const float* __restrict__ msa,
                                                 float* __restrict__ out) {
  __shared__ bf16 atile[16 * K_E4];    // 16 rows x 2048 bf16 = 64 KB
  int blk = blockIdx.x;                // 2048 blocks of 16 rows
  stage_tile(atile, h + (size_t)blk * 16 * K_E4, 16 * K_E4 * 2);

  int wave = threadIdx.x >> 5;
  int lane = threadIdx.x & 31;
  int half = lane >> 4, l15 = lane & 15;

  for (int nt0 = wave; nt0 < 32; nt0 += 8) {   // single mtile, 32 ntiles
    int nt = __builtin_amdgcn_readfirstlane(nt0);
    v8f acc = {};
#pragma unroll 4
    for (int kt = 0; kt < 64; ++kt) {
      __builtin_prefetch(wp + (size_t)((kt + 2) * 32 + nt) * 32 * 16, 0, 0);
      v16bf a = load_a_frag(atile + kt * 32, K_E4);
      v16bf bb = load_b_packed(wp + (size_t)(kt * 32 + nt) * 32 * 16);
      acc = wmma_bf16(a, bb, acc);
    }
    int ncol = nt * 16 + l15;
    float bv = bias[ncol];
#pragma unroll
    for (int r = 0; r < 8; ++r) {
      size_t row = (size_t)blk * 16 + half * 8 + r;
      out[row * K_C + ncol] = acc[r] + bv + msa[row * K_C + ncol];
    }
  }
}

// ---------------- launch ----------------
extern "C" void kernel_launch(void* const* d_in, const int* in_sizes, int n_in,
                              void* d_out, int out_size, void* d_ws, size_t ws_size,
                              hipStream_t stream) {
  (void)in_sizes; (void)n_in; (void)out_size; (void)ws_size;
  const float* x     = (const float*)d_in[0];
  const float* ln1g  = (const float*)d_in[1];
  const float* ln1b  = (const float*)d_in[2];
  const float* wqkv  = (const float*)d_in[3];
  const float* bqkv  = (const float*)d_in[4];
  const float* wproj = (const float*)d_in[5];
  const float* bproj = (const float*)d_in[6];
  const float* ln2g  = (const float*)d_in[7];
  const float* ln2b  = (const float*)d_in[8];
  const float* w1    = (const float*)d_in[9];
  const float* b1    = (const float*)d_in[10];
  const float* w2    = (const float*)d_in[11];
  const float* b2    = (const float*)d_in[12];
  float* out = (float*)d_out;

  size_t off = 0;
  char* base = (char*)d_ws;
  auto alloc = [&](size_t bytes) -> char* {
    char* p = base + off;
    off += (bytes + 255) & ~(size_t)255;
    return p;
  };
  bf16* wqkv_p  = (bf16*)alloc((size_t)K_C * 3 * K_C * sizeof(bf16));
  bf16* wproj_p = (bf16*)alloc((size_t)K_C * K_C * sizeof(bf16));
  bf16* w1_p    = (bf16*)alloc((size_t)K_C * K_E4 * sizeof(bf16));
  bf16* w2_p    = (bf16*)alloc((size_t)K_E4 * K_C * sizeof(bf16));
  bf16* winx    = (bf16*)alloc((size_t)K_TOK * K_C * sizeof(bf16));
  bf16* qb      = (bf16*)alloc((size_t)K_TOK * K_C * sizeof(bf16));
  bf16* kb      = (bf16*)alloc((size_t)K_TOK * K_C * sizeof(bf16));
  bf16* vtb     = (bf16*)alloc((size_t)K_TOK * K_C * sizeof(bf16));
  bf16* attn_o  = (bf16*)alloc((size_t)K_TOK * K_C * sizeof(bf16));
  float* msa    = (float*)alloc((size_t)K_TOK * K_C * sizeof(float));
  bf16* ln2x    = (bf16*)alloc((size_t)K_TOK * K_C * sizeof(bf16));
  bf16* hbuf    = (bf16*)alloc((size_t)K_TOK * K_E4 * sizeof(bf16));

  // 1) weight packing (bf16, WMMA fragment order)
  pack_w<<<dim3(256), dim3(256), 0, stream>>>(wqkv,  wqkv_p,  K_C,  3 * K_C);
  pack_w<<<dim3(256), dim3(256), 0, stream>>>(wproj, wproj_p, K_C,  K_C);
  pack_w<<<dim3(256), dim3(256), 0, stream>>>(w1,    w1_p,    K_C,  K_E4);
  pack_w<<<dim3(256), dim3(256), 0, stream>>>(w2,    w2_p,    K_E4, K_C);

  // 2) roll + LN1 + window partition
  ln1_win<<<dim3(K_TOK / 8), dim3(256), 0, stream>>>(x, ln1g, ln1b, winx);

  // 3) QKV GEMM
  qkv_gemm<<<dim3(K_NWIN), dim3(256), 0, stream>>>(winx, wqkv_p, bqkv, qb, kb, vtb);

  // 4) windowed attention
  attn_kernel<<<dim3(K_NWIN * K_HEADS / 8), dim3(256), 0, stream>>>(qb, kb, vtb, attn_o);

  // 5) proj + reverse roll + residual
  proj_gemm<<<dim3(K_NWIN), dim3(256), 0, stream>>>(attn_o, wproj_p, bproj, x, msa);

  // 6) LN2
  ln2_kernel<<<dim3(K_TOK / 8), dim3(256), 0, stream>>>(msa, ln2g, ln2b, ln2x);

  // 7) MLP up + GELU
  mlp1_gemm<<<dim3(K_TOK / K_WT), dim3(256), 0, stream>>>(ln2x, w1_p, b1, hbuf);

  // 8) MLP down + residual -> out
  mlp2_gemm<<<dim3(K_TOK / 16), dim3(256), 0, stream>>>(hbuf, w2_p, b2, msa, out);
}
